// ScaledDotProductAttention_37168646980165
// MI455X (gfx1250) — matrix operand order repaired and checked by reading, not compiled, not run on hardware
//
#include <hip/hip_runtime.h>

typedef __bf16 bf16_t;
typedef __attribute__((ext_vector_type(16))) __bf16 v16bf;
typedef __attribute__((ext_vector_type(8)))  float  v8f;

#define BATCH 16
#define NQ    4096
#define NK    4096
#define DHEAD 128
#define KPITCH 40    // 32 keys + 8 pad (bf16 units)
#define VPITCH 136   // 128 d + 8 pad (bf16 units)

union Frag16 { uint4 u[2]; v16bf v; };
union Pack4  { __bf16 h[4]; uint2 u; };

// ---------------- async global->LDS copy (CDNA5) with fallback ----------------
typedef int v4i __attribute__((vector_size(16)));
typedef __attribute__((address_space(1))) v4i gas_v4i;
typedef __attribute__((address_space(3))) v4i las_v4i;

#if __has_builtin(__builtin_amdgcn_global_load_async_to_lds_b128)
#define HAVE_ASYNC_LDS 1
__device__ __forceinline__ void async_copy16(void* dst, const void* src) {
  // flat global address == AS1 address; LDS offset == low 32 bits of flat shared address
  __builtin_amdgcn_global_load_async_to_lds_b128(
      (gas_v4i*)(unsigned long long)src,
      (las_v4i*)(unsigned)(unsigned long long)dst, 0, 0);
}
#else
#define HAVE_ASYNC_LDS 0
__device__ __forceinline__ void async_copy16(void* dst, const void* src) {
  *(uint4*)dst = *(const uint4*)src;
}
#endif

__device__ __forceinline__ void async_wait_keep8() {
#if HAVE_ASYNC_LDS
  asm volatile("s_wait_asynccnt 0x8" ::: "memory");   // keep prefetch batch in flight
#endif
}
__device__ __forceinline__ void async_wait_all() {
#if HAVE_ASYNC_LDS
  asm volatile("s_wait_asynccnt 0x0" ::: "memory");
#endif
}

// ---------- preprocessing: fp32 -> bf16 ----------
__global__ void cvt_scale_bf16(const float* __restrict__ src, bf16_t* __restrict__ dst,
                               int n4, float scale) {
  int i = blockIdx.x * blockDim.x + threadIdx.x;
  if (i >= n4) return;
  const float4 f = ((const float4*)src)[i];
  Pack4 p;
  p.h[0] = (__bf16)(f.x * scale);
  p.h[1] = (__bf16)(f.y * scale);
  p.h[2] = (__bf16)(f.z * scale);
  p.h[3] = (__bf16)(f.w * scale);
  ((uint2*)dst)[i] = p.u;
}

// K: convert + transpose [B][Nk][d] -> [B][d][Nk]
__global__ void cvt_transpose_bf16(const float* __restrict__ src, bf16_t* __restrict__ dst,
                                   int n4) {
  int i = blockIdx.x * blockDim.x + threadIdx.x;
  if (i >= n4) return;
  const size_t o = (size_t)i * 4;
  const int b  = (int)(o / ((size_t)DHEAD * NK));
  const int r  = (int)(o % ((size_t)DHEAD * NK));
  const int dd = r / NK;
  const int n0 = r % NK;
  const float* s = src + ((size_t)b * NK) * DHEAD + dd;
  Pack4 p;
  p.h[0] = (__bf16)s[(size_t)(n0 + 0) * DHEAD];
  p.h[1] = (__bf16)s[(size_t)(n0 + 1) * DHEAD];
  p.h[2] = (__bf16)s[(size_t)(n0 + 2) * DHEAD];
  p.h[3] = (__bf16)s[(size_t)(n0 + 3) * DHEAD];
  ((uint2*)dst)[i] = p.u;
}

// cooperative stage of one 32-key tile (K^T slab 128x32 + V slab 32x128) into LDS
__device__ __forceinline__ void stage_tile(const bf16_t* __restrict__ KtB,
                                           const bf16_t* __restrict__ VbB,
                                           bf16_t* kdst, bf16_t* vdst,
                                           int kbase, int t) {
  // K^T: thread t copies d-row t (64 bytes, 4 x b128)
  const bf16_t* ksrc = KtB + (size_t)t * NK + kbase;
  bf16_t* kd = kdst + t * KPITCH;
#pragma unroll
  for (int i = 0; i < 4; ++i) async_copy16(kd + i * 8, ksrc + i * 8);
  // V: thread t copies quarter of key-row (t>>2), 64 bytes
  const int vr = t >> 2, cg = t & 3;
  const bf16_t* vsrc = VbB + ((size_t)(kbase + vr)) * DHEAD + cg * 32;
  bf16_t* vd = vdst + vr * VPITCH + cg * 32;
#pragma unroll
  for (int i = 0; i < 4; ++i) async_copy16(vd + i * 8, vsrc + i * 8);
}

// ---------------- flash attention main kernel ----------------
// 128 threads = 4 waves; waves own 4 consecutive 16-row q-tiles of one batch and
// share a double-buffered K/V key-tile stream staged in LDS.
__global__ __launch_bounds__(128, 1)
void fa_causal_bf16_wmma(const bf16_t* __restrict__ Qb,   // [B][Nq][d], pre-scaled
                         const bf16_t* __restrict__ Kt,   // [B][d][Nk]
                         const bf16_t* __restrict__ Vb,   // [B][Nk][d]
                         float* __restrict__ Out) {       // [B][Nq][d] f32
  __shared__ bf16_t kbuf[2][128 * KPITCH];
  __shared__ bf16_t vbuf[2][32 * VPITCH];
  __shared__ bf16_t pshared[4][16 * 32];

  const int tid  = threadIdx.x;
  const int lane = tid & 31;
  const int wave = tid >> 5;
  const int tile = blockIdx.x * 4 + wave;    // batch boundary never splits a block
  const int b    = tile >> 8;                // Nq/16 = 256 q-tiles per batch
  const int qb   = (tile & 255) * 16;

  const int lg  = lane >> 4;
  const int ln  = lane & 15;
  const int kb8 = lg * 8;

  const bf16_t* KtB = Kt + (size_t)b * DHEAD * NK;
  const bf16_t* VbB = Vb + (size_t)b * NK * DHEAD;

  // ---- Q A-fragments, resident ----
  v16bf qa[4];
  {
    const bf16_t* qrow = Qb + (((size_t)b * NQ) + qb + ln) * DHEAD;
#pragma unroll
    for (int c = 0; c < 4; ++c) {
      Frag16 f;
      f.u[0] = *(const uint4*)(qrow + c * 32 + kb8);
      f.u[1] = *(const uint4*)(qrow + c * 32 + kb8 + 16);
      qa[c] = f.v;
    }
  }

  float m_i[8], l_i[8];
  v8f acc[8] = {};
#pragma unroll
  for (int r = 0; r < 8; ++r) { m_i[r] = -__builtin_inff(); l_i[r] = 0.0f; }

  const int nktMine = (qb + 47) / 32;                          // my causal bound
  const int nktMax  = ((((blockIdx.x * 4 + 3) & 255) * 16) + 47) / 32; // block max

  stage_tile(KtB, VbB, kbuf[0], vbuf[0], 0, tid);

  for (int kt = 0; kt < nktMax; ++kt) {
    const int buf   = kt & 1;
    const int kbase = kt * 32;
    const bool more = (kt + 1) < nktMax;

    if (more) {
      stage_tile(KtB, VbB, kbuf[buf ^ 1], vbuf[buf ^ 1], kbase + 32, tid);
      async_wait_keep8();
    } else {
      async_wait_all();
    }
    __syncthreads();

    if (kt < nktMine) {
      // ---- S = Q * K^T from LDS ----
      v8f s0 = {}, s1 = {};
#pragma unroll
      for (int c = 0; c < 4; ++c) {
        const bf16_t* kr = kbuf[buf] + (c * 32) * KPITCH + ln;
        Frag16 b0, b1;
#pragma unroll
        for (int s = 0; s < 16; ++s) {
          const int kk = (s < 8) ? (kb8 + s) : (16 + kb8 + s - 8);
          b0.v[s] = kr[kk * KPITCH];
          b1.v[s] = kr[kk * KPITCH + 16];
        }
        s0 = __builtin_amdgcn_wmma_f32_16x16x32_bf16(false, qa[c], false, b0.v,
                                                     (short)0, s0, false, false);
        s1 = __builtin_amdgcn_wmma_f32_16x16x32_bf16(false, qa[c], false, b1.v,
                                                     (short)0, s1, false, false);
      }

      // ---- causal mask (boundary tile only) ----
      if (kbase + 31 > qb) {
        const int k0 = kbase + ln;
#pragma unroll
        for (int r = 0; r < 8; ++r) {
          const int q = qb + r + 8 * lg;
          if (k0 > q)      s0[r] = -__builtin_inff();
          if (k0 + 16 > q) s1[r] = -__builtin_inff();
        }
      }

      // ---- online softmax ----
      float alpha[8];
#pragma unroll
      for (int r = 0; r < 8; ++r) {
        float mx = fmaxf(s0[r], s1[r]);
        mx = fmaxf(mx, __shfl_xor(mx, 1, 32));
        mx = fmaxf(mx, __shfl_xor(mx, 2, 32));
        mx = fmaxf(mx, __shfl_xor(mx, 4, 32));
        mx = fmaxf(mx, __shfl_xor(mx, 8, 32));
        const float mn = fmaxf(m_i[r], mx);
        alpha[r] = __expf(m_i[r] - mn);
        m_i[r] = mn;
        const float p0 = __expf(s0[r] - mn);
        const float p1 = __expf(s1[r] - mn);
        s0[r] = p0; s1[r] = p1;
        float rs = p0 + p1;
        rs += __shfl_xor(rs, 1, 32);
        rs += __shfl_xor(rs, 2, 32);
        rs += __shfl_xor(rs, 4, 32);
        rs += __shfl_xor(rs, 8, 32);
        l_i[r] = l_i[r] * alpha[r] + rs;
      }

      // ---- P: C-layout -> A-layout via per-wave LDS tile ----
      bf16_t* ps = pshared[wave];
#pragma unroll
      for (int r = 0; r < 8; ++r) {
        const int row = r + 8 * lg;
        ps[row * 32 + ln]      = (__bf16)s0[r];
        ps[row * 32 + 16 + ln] = (__bf16)s1[r];
      }
      asm volatile("s_wait_dscnt 0" ::: "memory");
      Frag16 pf;
      pf.u[0] = *(const uint4*)(ps + ln * 32 + kb8);
      pf.u[1] = *(const uint4*)(ps + ln * 32 + kb8 + 16);

      // ---- O = alpha*O + P * V (V from LDS) ----
#pragma unroll
      for (int n = 0; n < 8; ++n) {
        v8f a = acc[n];
#pragma unroll
        for (int r = 0; r < 8; ++r) a[r] *= alpha[r];
        const bf16_t* vr = vbuf[buf] + n * 16 + ln;
        Frag16 vf;
#pragma unroll
        for (int s = 0; s < 16; ++s) {
          const int kk = (s < 8) ? (kb8 + s) : (16 + kb8 + s - 8);
          vf.v[s] = vr[kk * VPITCH];
        }
        acc[n] = __builtin_amdgcn_wmma_f32_16x16x32_bf16(false, pf.v, false, vf.v,
                                                         (short)0, a, false, false);
      }
    }
    __syncthreads();   // protect buffer reuse next iteration
  }

  // ---- epilogue ----
  float inv[8];
#pragma unroll
  for (int r = 0; r < 8; ++r) inv[r] = 1.0f / l_i[r];
#pragma unroll
  for (int n = 0; n < 8; ++n) {
#pragma unroll
    for (int r = 0; r < 8; ++r) {
      const int row = r + 8 * lg;
      Out[((size_t)b * NQ + qb + row) * DHEAD + n * 16 + ln] = acc[n][r] * inv[r];
    }
  }
}

extern "C" void kernel_launch(void* const* d_in, const int* in_sizes, int n_in,
                              void* d_out, int out_size, void* d_ws, size_t ws_size,
                              hipStream_t stream) {
  const float* q = (const float*)d_in[0];
  const float* k = (const float*)d_in[1];
  const float* v = (const float*)d_in[2];
  float* out = (float*)d_out;

  const size_t nElem = (size_t)BATCH * NQ * DHEAD;
  bf16_t* qbf = (bf16_t*)d_ws;
  bf16_t* ktb = qbf + nElem;
  bf16_t* vbf = ktb + (size_t)BATCH * DHEAD * NK;

  const int n4 = (int)(nElem / 4);
  const int cvtBlocks = (n4 + 255) / 256;
  const float scale = 0.088388347648318447f;  // 1/sqrt(128)

  cvt_scale_bf16<<<cvtBlocks, 256, 0, stream>>>(q, qbf, n4, scale);
  cvt_transpose_bf16<<<cvtBlocks, 256, 0, stream>>>(k, ktb, n4);
  cvt_scale_bf16<<<cvtBlocks, 256, 0, stream>>>(v, vbf, n4, 1.0f);

  fa_causal_bf16_wmma<<<1024, 128, 0, stream>>>(qbf, ktb, vbf, out);
}
